// SparseAttention3D_8392366096919
// MI455X (gfx1250) — matrix operand order repaired and checked by reading, not compiled, not run on hardware
//
#include <hip/hip_runtime.h>
#include <hip/hip_bf16.h>

// ---------------------------------------------------------------------------
// SparseAttention3D for MI455X (gfx1250, wave32, WMMA + TDM)
//   B=2, C=128, N=D*H*W=4096, heads=4, hd=32, TOPK=256, GROUPS=32
// - All matmuls: v_wmma_f32_16x16x32_f16 (f16 in, fp32 accumulate)
// - GEMM B-operand tiles staged into LDS by the Tensor Data Mover
//   (tensor_load_to_lds + s_wait_tensorcnt) instead of stride-4096 gathers
// - Top-K: 16x4096 fp32 score slab fully resident in 256KB dynamic LDS
//   (CDNA5 320KB/WGP), 24-step bisection threshold + ballot-driven AV gather
// ---------------------------------------------------------------------------

typedef __attribute__((ext_vector_type(16))) _Float16     v16h;
typedef __attribute__((ext_vector_type(8)))  float        v8f;
typedef __attribute__((ext_vector_type(4)))  unsigned int v4u;
typedef __attribute__((ext_vector_type(8)))  int          v8i;
typedef __attribute__((ext_vector_type(4)))  int          v4i;

#define CC       128
#define NN       4096
#define NH       4
#define HD       32
#define TOPK_K   256
#define GROUPS_G 32
#define SCALE    0.17677669529663687f   /* 32^-0.5 */

#if __has_builtin(__builtin_amdgcn_tensor_load_to_lds) && \
    __has_builtin(__builtin_amdgcn_s_wait_tensorcnt)
#define HAVE_TDM 1
#else
#define HAVE_TDM 0
#endif

__device__ __forceinline__ float wredSum(float v) {
#pragma unroll
    for (int o = 16; o > 0; o >>= 1) v += __shfl_xor(v, o, 32);
    return v;
}
__device__ __forceinline__ float wredMax(float v) {
#pragma unroll
    for (int o = 16; o > 0; o >>= 1) v = fmaxf(v, __shfl_xor(v, o, 32));
    return v;
}
__device__ __forceinline__ int wredSumI(int v) {
#pragma unroll
    for (int o = 16; o > 0; o >>= 1) v += __shfl_xor(v, o, 32);
    return v;
}

#if HAVE_TDM
// Issue a 2-D TDM tile load: tile_d1 rows x tile_d0 elements into LDS.
// D# layout per CDNA5 ISA (§8.3 group0, §8.4 group1).
// data_size code: 0=1B 1=2B 2=4B 3=8B. Strides/dims in data_size units.
__device__ __forceinline__ void tdm_load_2d(unsigned int lds_off,
                                            const void* gaddr,
                                            unsigned int tensor_d0,
                                            unsigned int tensor_d1,
                                            unsigned int tile_d0,
                                            unsigned int tile_d1,
                                            unsigned int stride0,
                                            unsigned int dsz_code) {
    unsigned long long ga = (unsigned long long)(uintptr_t)gaddr;
    v4u g0;
    g0.x = 1u;                                         // count=1, user mode
    g0.y = lds_off;                                    // lds_addr (bytes)
    g0.z = (unsigned int)(ga & 0xffffffffull);         // global_addr[31:0]
    g0.w = (unsigned int)((ga >> 32) & 0x01ffffffull)  // global_addr[56:32]
           | (2u << 30);                               // type=2 ("image")
    v8i g1;
    g1[0] = (int)(dsz_code << 16);                     // wg_mask=0, data_size
    g1[1] = (int)((tensor_d0 & 0xffffu) << 16);        // td0[15:0] @ [63:48]
    g1[2] = (int)((tensor_d0 >> 16) & 0xffffu)         // td0[31:16]
          | (int)((tensor_d1 & 0xffffu) << 16);        // td1[15:0]
    g1[3] = (int)((tensor_d1 >> 16) & 0xffffu)         // td1[31:16]
          | (int)((tile_d0 & 0xffffu) << 16);          // tile_dim0
    g1[4] = (int)(tile_d1 & 0xffffu);                  // tile_dim1, tile_dim2=0
    g1[5] = (int)stride0;                              // dim0_stride[31:0]
    g1[6] = 0;                                         // dim0_stride[47:32]=0
    g1[7] = 0;                                         // dim1_stride unused(2D)
    v4i gz = {0, 0, 0, 0};
#if defined(__clang_major__) && (__clang_major__ >= 23)
    v8i gz8 = {0, 0, 0, 0, 0, 0, 0, 0};
    __builtin_amdgcn_tensor_load_to_lds(g0, g1, gz, gz, gz8, 0);
#else
    __builtin_amdgcn_tensor_load_to_lds(g0, g1, gz, gz, 0);
#endif
}
#endif

// ---------------------------------------------------------------------------
// Kernel 1: per-(batch,group) mean / inv-std.  64 blocks x 256 threads.
// ---------------------------------------------------------------------------
__global__ void gn_stats_kernel(const float* __restrict__ x,
                                float* __restrict__ stats) {
    const int bg = blockIdx.x;                 // b*32 + g
    const int b  = bg >> 5, g = bg & 31;
    const float* base = x + (((size_t)b * CC) + (size_t)g * 4) * NN;
    float s = 0.f, s2 = 0.f;
    for (int i = threadIdx.x; i < 4 * NN; i += blockDim.x) {
        float v = base[i];
        s += v; s2 += v * v;
    }
    __shared__ float a1[256], a2[256];
    a1[threadIdx.x] = s; a2[threadIdx.x] = s2;
    __syncthreads();
#pragma unroll
    for (int st = 128; st > 0; st >>= 1) {
        if ((int)threadIdx.x < st) {
            a1[threadIdx.x] += a1[threadIdx.x + st];
            a2[threadIdx.x] += a2[threadIdx.x + st];
        }
        __syncthreads();
    }
    if (threadIdx.x == 0) {
        float mu  = a1[0] * (1.f / 16384.f);
        float var = a2[0] * (1.f / 16384.f) - mu * mu;
        stats[bg * 2 + 0] = mu;
        stats[bg * 2 + 1] = rsqrtf(var + 1e-5f);
    }
}

// ---------------------------------------------------------------------------
// Kernel 2: QKV GEMM, GroupNorm fused. M=384(o), K=128(c), N=4096(n).
// TDM stages the 128x64 x-panel (fp32, 32KB) into LDS once per block;
// WMMA B operands (normalized, f16) are built from LDS reads.
// ---------------------------------------------------------------------------
__global__ void qkv_wmma_kernel(const float* __restrict__ x,
                                const float* __restrict__ gnw,
                                const float* __restrict__ gnb,
                                const float* __restrict__ qkvw,
                                const float* __restrict__ qkvb,
                                const float* __restrict__ stats,
                                _Float16* __restrict__ Qb,
                                _Float16* __restrict__ Kb,
                                _Float16* __restrict__ Vb) {
    __shared__ float Xsm[CC * 64];                      // 32 KB: [c][n']
    const int b      = blockIdx.z;
    const int mTile  = blockIdx.y;                      // 0..23
    const int lane   = threadIdx.x & 31;
    const int wave   = threadIdx.x >> 5;                // 0..3
    const int nBase0 = blockIdx.x * 64;
    const int col    = lane & 15;
    const int hl     = lane >> 4;
    const int nLoc   = wave * 16 + col;
    const int n      = nBase0 + nLoc;
    const int oRow   = mTile * 16 + col;

    // ---- stage x[b, 0..127, nBase0..+63] into LDS ----
    const float* xpan = x + (size_t)b * CC * NN + nBase0;
#if HAVE_TDM
    if (threadIdx.x < 32) {                             // wave 0 issues DMA
        tdm_load_2d((unsigned int)(uintptr_t)(void*)Xsm, xpan,
                    /*tensor_d0=*/NN, /*tensor_d1=*/CC,
                    /*tile_d0=*/64, /*tile_d1=*/CC,
                    /*stride0=*/NN, /*dsz=*/2u);
        __builtin_amdgcn_s_wait_tensorcnt(0);
    }
#else
    for (int i = threadIdx.x; i < CC * 64; i += blockDim.x)
        Xsm[i] = xpan[(size_t)(i >> 6) * NN + (i & 63)];
#endif
    __syncthreads();

    v8f acc = {};
#pragma unroll
    for (int ks = 0; ks < CC; ks += 32) {
        v16h A, Bm;
        // A 16x32 f16 lane layout: lanes 0-15 K={0..7,16..23}, lanes 16-31 +8
#pragma unroll
        for (int e = 0; e < 8; ++e) {
            A[e]     = (_Float16)qkvw[oRow * CC + ks + hl * 8 + e];
            A[e + 8] = (_Float16)qkvw[oRow * CC + ks + 16 + hl * 8 + e];
        }
        // B 32x16 from LDS, GroupNorm applied on the fly
#pragma unroll
        for (int e = 0; e < 16; ++e) {
            int c = ks + (e < 8 ? hl * 8 + e : 16 + hl * 8 + (e - 8));
            int g = c >> 2;
            float mu = stats[(b * GROUPS_G + g) * 2 + 0];
            float rs = stats[(b * GROUPS_G + g) * 2 + 1];
            float xv = Xsm[c * 64 + nLoc];
            Bm[e] = (_Float16)(((xv - mu) * rs) * gnw[c] + gnb[c]);
        }
        acc = __builtin_amdgcn_wmma_f32_16x16x32_f16(
            false, A, false, Bm, (short)0, acc, false, false);
    }
#pragma unroll
    for (int v = 0; v < 8; ++v) {
        int o = mTile * 16 + v + 8 * hl;                // D row
        float val = acc[v] + qkvb[o];
        int which = o >> 7;                             // 0=q 1=k 2=v
        int cc    = o & 127;
        int head  = cc >> 5, d = cc & 31;
        size_t idx = (((size_t)b * NH + head) * NN + n) * HD + d;
        if (which == 0)      Qb[idx] = (_Float16)(val * SCALE);
        else if (which == 1) Kb[idx] = (_Float16)val;
        else                 Vb[idx] = (_Float16)val;
    }
}

// ---------------------------------------------------------------------------
// Kernel 3: attention with exact-ish top-256 (unchanged WMMA/LDS design).
// ---------------------------------------------------------------------------
__global__ void attn_topk_kernel(const _Float16* __restrict__ Qb,
                                 const _Float16* __restrict__ Kb,
                                 const _Float16* __restrict__ Vb,
                                 _Float16* __restrict__ outb) {
    extern __shared__ float sm[];                       // 16 * 4096 floats
    const int b     = blockIdx.z;
    const int head  = blockIdx.y;
    const int qBase = blockIdx.x * 16;
    const int lane  = threadIdx.x & 31;
    const int wave  = threadIdx.x >> 5;                 // 0..15
    const int col   = lane & 15;
    const int hl    = lane >> 4;

    const size_t hbase = ((size_t)b * NH + head) * NN * HD;
    const _Float16* Q = Qb + hbase;
    const _Float16* K = Kb + hbase;
    const _Float16* V = Vb + hbase;

    // ---- Phase 1: scores = Qs @ K^T, one wave covers 256 keys -------------
    v16h A;
    {
        const _Float16* qr = Q + (size_t)(qBase + col) * HD;
#pragma unroll
        for (int e = 0; e < 8; ++e) {
            A[e]     = qr[hl * 16 + e];
            A[e + 8] = qr[hl * 16 + 8 + e];
        }
    }
#pragma unroll
    for (int t = 0; t < 16; ++t) {
        const int keyT = (wave * 16 + t) * 16;
        v16h Bm;
        const _Float16* kr = K + (size_t)(keyT + col) * HD + hl * 16;
#pragma unroll
        for (int e = 0; e < 16; ++e) Bm[e] = kr[e];
        v8f c = {};
        c = __builtin_amdgcn_wmma_f32_16x16x32_f16(
            false, A, false, Bm, (short)0, c, false, false);
#pragma unroll
        for (int v = 0; v < 8; ++v)
            sm[(size_t)(v + 8 * hl) * NN + keyT + col] = c[v];
    }
    __syncthreads();

    // ---- Phase 2: wave w owns query row w ---------------------------------
    float* srow = sm + (size_t)wave * NN;
    float mx = -3.4e38f, mn = 3.4e38f;
    for (int i = lane; i < NN; i += 32) {
        float s = srow[i];
        mx = fmaxf(mx, s); mn = fminf(mn, s);
    }
    mx = wredMax(mx);
    mn = -wredMax(-mn);

    float lo = mn, hi = mx;
#pragma unroll 1
    for (int it = 0; it < 24; ++it) {
        float mid = 0.5f * (lo + hi);
        int cnt = 0;
        for (int i = lane; i < NN; i += 32) cnt += (srow[i] >= mid) ? 1 : 0;
        cnt = wredSumI(cnt);
        if (cnt > TOPK_K) lo = mid; else hi = mid;
    }
    const float thr = hi;   // count(>= thr) <= 256, within fp32 bisection

    // ---- Phase 3: softmax over selected + AV gather (lane = head dim) -----
    float acc = 0.f, dsum = 0.f;
#pragma unroll 1
    for (int kb = 0; kb < NN; kb += 32) {
        if (kb + 32 < NN)
            __builtin_prefetch(V + (size_t)(kb + 32) * HD + lane, 0, 1);
        float s  = srow[kb + lane];
        bool sel = (s >= thr);
        float p  = sel ? __expf(s - mx) : 0.f;
        dsum += p;
        unsigned mask = (unsigned)__ballot(sel);
        while (mask) {
            int k0 = __builtin_ctz(mask);
            mask &= mask - 1;
            float pk = __shfl(p, k0, 32);
            acc += pk * (float)V[(size_t)(kb + k0) * HD + lane];
        }
    }
    dsum = wredSum(dsum);
    // out layout [b][c][n], c = head*32 + d  (d = lane)
    outb[((size_t)b * CC + head * HD + lane) * NN + (qBase + wave)] =
        (_Float16)(acc / dsum);
}

// ---------------------------------------------------------------------------
// Kernel 4: projection GEMM + bias + residual.
// TDM stages the 128x64 f16 attn-output panel (16KB) into LDS per block.
// ---------------------------------------------------------------------------
__global__ void proj_wmma_kernel(const float* __restrict__ x,
                                 const float* __restrict__ projw,
                                 const float* __restrict__ projb,
                                 const _Float16* __restrict__ outbuf,
                                 float* __restrict__ y) {
    __shared__ _Float16 Bsm[CC * 64];                   // 16 KB: [c][n']
    const int b      = blockIdx.z;
    const int mTile  = blockIdx.y;                      // 0..7
    const int lane   = threadIdx.x & 31;
    const int wave   = threadIdx.x >> 5;                // 0..3
    const int nBase0 = blockIdx.x * 64;
    const int col    = lane & 15;
    const int hl     = lane >> 4;
    const int nLoc   = wave * 16 + col;
    const int n      = nBase0 + nLoc;
    const int oRow   = mTile * 16 + col;

    const _Float16* opan = outbuf + (size_t)b * CC * NN + nBase0;
#if HAVE_TDM
    if (threadIdx.x < 32) {
        tdm_load_2d((unsigned int)(uintptr_t)(void*)Bsm, opan,
                    /*tensor_d0=*/NN, /*tensor_d1=*/CC,
                    /*tile_d0=*/64, /*tile_d1=*/CC,
                    /*stride0=*/NN, /*dsz=*/1u);
        __builtin_amdgcn_s_wait_tensorcnt(0);
    }
#else
    for (int i = threadIdx.x; i < CC * 64; i += blockDim.x)
        Bsm[i] = opan[(size_t)(i >> 6) * NN + (i & 63)];
#endif
    __syncthreads();

    v8f acc = {};
#pragma unroll
    for (int ks = 0; ks < CC; ks += 32) {
        v16h A, Bm;
#pragma unroll
        for (int e = 0; e < 8; ++e) {
            A[e]     = (_Float16)projw[oRow * CC + ks + hl * 8 + e];
            A[e + 8] = (_Float16)projw[oRow * CC + ks + 16 + hl * 8 + e];
        }
#pragma unroll
        for (int e = 0; e < 16; ++e)
            Bm[e] = Bsm[(ks + (e < 8 ? hl * 8 + e : 16 + hl * 8 + (e - 8))) * 64 + nLoc];
        acc = __builtin_amdgcn_wmma_f32_16x16x32_f16(
            false, A, false, Bm, (short)0, acc, false, false);
    }
#pragma unroll
    for (int v = 0; v < 8; ++v) {
        int o = mTile * 16 + v + 8 * hl;
        size_t idx = ((size_t)b * CC + o) * NN + n;
        y[idx] = x[idx] + acc[v] + projb[o];
    }
}

// ---------------------------------------------------------------------------
extern "C" void kernel_launch(void* const* d_in, const int* in_sizes, int n_in,
                              void* d_out, int out_size, void* d_ws,
                              size_t ws_size, hipStream_t stream) {
    (void)in_sizes; (void)n_in; (void)out_size; (void)ws_size;
    const float* x     = (const float*)d_in[0];
    const float* gnw   = (const float*)d_in[1];
    const float* gnb   = (const float*)d_in[2];
    const float* qkvw  = (const float*)d_in[3];
    const float* qkvb  = (const float*)d_in[4];
    const float* projw = (const float*)d_in[5];
    const float* projb = (const float*)d_in[6];
    float*       out   = (float*)d_out;

    char* w = (char*)d_ws;
    float*    stats = (float*)w;                        // 512 B (pad to 1KB)
    _Float16* Qb    = (_Float16*)(w + 1024);
    const size_t qkvElems = (size_t)2 * NH * NN * HD;   // 1,048,576 halfs
    _Float16* Kb    = Qb + qkvElems;
    _Float16* Vb    = Kb + qkvElems;
    _Float16* outb  = Vb + qkvElems;                    // [b][c][n] f16

    gn_stats_kernel<<<2 * GROUPS_G, 256, 0, stream>>>(x, stats);
    qkv_wmma_kernel<<<dim3(NN / 64, 24, 2), 128, 0, stream>>>(
        x, gnw, gnb, qkvw, qkvb, stats, Qb, Kb, Vb);
    attn_topk_kernel<<<dim3(NN / 16, NH, 2), 512, 16 * NN * sizeof(float),
                       stream>>>(Qb, Kb, Vb, outb);
    proj_wmma_kernel<<<dim3(NN / 64, 8, 2), 128, 0, stream>>>(
        x, projw, projb, outb, out);
}
